// SchNetPotential_84851373899788
// MI455X (gfx1250) — matrix operand order repaired and checked by reading, not compiled, not run on hardware
//
#include <hip/hip_runtime.h>
#include <hip/hip_bf16.h>
#include <stdint.h>

// ---------------- problem constants (from reference) ----------------
#define HID   128
#define NF    128
#define NG    50
#define NGP   64          // NG padded to WMMA-friendly K
#define NI    3
#define NB    4
#define NA    512
#define CUTOFF_F 5.0f
#define GWIDTH  (CUTOFF_F / 50.0f)   // width = cutoff/num_gaussians
#define GSTEP   (CUTOFF_F / 49.0f)   // linspace(0, cutoff, 50) step

typedef __attribute__((ext_vector_type(16))) __bf16 bf16x16;
typedef __attribute__((ext_vector_type(8)))  float  f32x8;
typedef __attribute__((ext_vector_type(4)))  unsigned int u32x4;
typedef __attribute__((ext_vector_type(8)))  int i32x8;
typedef __attribute__((ext_vector_type(4)))  int i32x4;

#if __has_builtin(__builtin_amdgcn_tensor_load_to_lds)
#define HAS_TDM 1
#else
#define HAS_TDM 0
#endif

union Bf16Tile { uint4 q[2]; bf16x16 v; };
union Bf16Bits { unsigned short s; __bf16 b; };

__device__ __forceinline__ unsigned short bf16_bits(float f) {
  union { float f; uint32_t u; } c; c.f = f;
  uint32_t u = c.u;
  return (unsigned short)((u + 0x7FFFu + ((u >> 16) & 1u)) >> 16); // RNE
}

__device__ __forceinline__ __bf16 to_bf16(float f) {
  Bf16Bits o; o.s = bf16_bits(f);
  return o.b;
}

__device__ __forceinline__ float softplusf(float x) {
  return fmaxf(x, 0.0f) + log1pf(__expf(-fabsf(x)));
}

// A-matrix (16x32, MxK, bf16) fragment load from row-major LDS.
// lane = m + 16*s ; VGPR v, half h covers k = (v>=4?16:0) + s*8 + (v&3)*2 + h
// -> two contiguous 16B chunks per lane.
__device__ __forceinline__ bf16x16 load_A_16x32(const __bf16* base,
                                                int rowStrideHalf, int k0, int lane) {
  int m = lane & 15, s = lane >> 4;
  const char* p = (const char*)base + (size_t)(m * rowStrideHalf + k0 + s * 8) * 2;
  Bf16Tile t;
  t.q[0] = *(const uint4*)(p);
  t.q[1] = *(const uint4*)(p + 32);   // +16 halves (second K group)
  return t.v;
}

// B-matrix (32x16, KxN, bf16) fragment load from row-major LDS.
// lane l holds N = n0 + l%16; element s covers K = k0 + 8*(l/16) + (s&7) + (s>=8?16:0).
__device__ __forceinline__ bf16x16 load_B_32x16(const __bf16* base,
                                                int rowStrideHalf, int k0, int n0, int lane) {
  int n = lane & 15, h = lane >> 4;
  bf16x16 v;
#pragma unroll
  for (int s = 0; s < 16; ++s) {
    int k = k0 + 8 * h + (s & 7) + (s >= 8 ? 16 : 0);
    v[s] = base[k * rowStrideHalf + n0 + n];
  }
  return v;
}

__device__ __forceinline__ f32x8 wmma_bf16(bf16x16 a, bf16x16 b, f32x8 c) {
  return __builtin_amdgcn_wmma_f32_16x16x32_bf16(false, a, false, b,
                                                 (short)0, c, false, false);
}

// ---- TDM: DMA a contiguous byte range from global into LDS (1-row tensor) ----
// D# per CDNA5 ISA §8.3/8.4: group0 = {flags, lds_addr, global_addr lo/hi|type},
// group1 = {data_size=8B, tensor_dim0 = tile_dim0 = nbytes/8, dim1 = 1}.
__device__ __forceinline__ void tdm_load_bytes(const void* gsrc, void* ldst,
                                               uint32_t nbytes) {
#if HAS_TDM
  uint32_t lds = (uint32_t)(uintptr_t)ldst;      // LDS aperture: addr[31:0] = byte offset
  uint64_t ga  = (uint64_t)(uintptr_t)gsrc;
  uint32_t n8  = nbytes >> 3;                    // 8-byte elements
  u32x4 g0 = {0u, 0u, 0u, 0u};
  g0[0] = 1u;                                     // count=1 (valid user D#)
  g0[1] = lds;                                    // lds_addr (bits 63:32)
  g0[2] = (uint32_t)ga;                           // global_addr[31:0]
  g0[3] = (uint32_t)(ga >> 32) | 0x80000000u;     // global_addr[56:32] | type=2
  i32x8 g1 = {0, 0, 0, 0, 0, 0, 0, 0};
  g1[0] = 0x00030000;                             // data_size=3 (8 bytes)
  g1[1] = (int)(n8 << 16);                        // tensor_dim0 low16 (bits 79:64..)
  g1[2] = 0x00010000;                             // tensor_dim0 hi=0, tensor_dim1=1
  g1[3] = (int)(n8 << 16);                        // tile_dim0 (bits 127:112)
  g1[4] = 1;                                      // tile_dim1=1, tile_dim2=0
  g1[5] = (int)n8;                                // tensor_dim0_stride low32
  i32x4 g2 = {0, 0, 0, 0};
  i32x4 g3 = {0, 0, 0, 0};
#if defined(__clang_major__) && (__clang_major__ >= 23)
  i32x8 g4 = {0, 0, 0, 0, 0, 0, 0, 0};
  __builtin_amdgcn_tensor_load_to_lds(g0, g1, g2, g3, g4, 0);
#else
  __builtin_amdgcn_tensor_load_to_lds(g0, g1, g2, g3, 0);
#endif
#else
  (void)gsrc; (void)ldst; (void)nbytes;
#endif
}

__device__ __forceinline__ void wait_tensorcnt0() {
#if __has_builtin(__builtin_amdgcn_s_wait_tensorcnt)
  __builtin_amdgcn_s_wait_tensorcnt(0);
#else
  asm volatile("s_wait_tensorcnt 0x0" ::: "memory");
#endif
}

// ---------------- embedding lookup: feats[b,n,:] = emb[idx[b,n]] ----------------
__global__ void embed_kernel(const float* __restrict__ emb,
                             const int* __restrict__ idxs,
                             float* __restrict__ feats, int total) {
  int g = blockIdx.x * 256 + threadIdx.x;
  if (g >= total) return;
  int f = g & (HID - 1);
  int bn = g >> 7;
  feats[g] = emb[idxs[bn] * HID + f];
}

// ---- one-time weight conversion: rbf_w1 (NG->64 padded) + rbf_w2 -> bf16 ----
__global__ void pack_weights_kernel(const float* __restrict__ rbf_w1,
                                    const float* __restrict__ rbf_w2,
                                    __bf16* __restrict__ w1bf,
                                    __bf16* __restrict__ w2bf) {
  int g = blockIdx.x * 256 + threadIdx.x;
  const int totalW1 = NI * NGP * NF;
  const int totalW2 = NI * NF * NF;
  if (g < totalW1) {
    int it = g / (NGP * NF);
    int r  = g % (NGP * NF);
    int row = r >> 7, f = r & 127;
    w1bf[g] = to_bf16(row < NG ? rbf_w1[((size_t)it * NG + row) * NF + f] : 0.0f);
  } else if (g < totalW1 + totalW2) {
    int x = g - totalW1;
    w2bf[x] = to_bf16(rbf_w2[x]);
  }
}

// ---------------- generic M x 128 x 128 bf16-WMMA GEMM ----------------
// Out = act(A @ W + bias) [+ resid]; act: 0=none, 1=softplus
__global__ void __launch_bounds__(256)
gemm128_kernel(const float* __restrict__ A, const float* __restrict__ W,
               const float* __restrict__ bias, const float* __restrict__ resid,
               float* __restrict__ Out, int M, int act) {
  extern __shared__ char smem[];
  __bf16* Ws = (__bf16*)smem;              // 128*128 bf16
  __bf16* As = Ws + 128 * 128;             // 128*128 bf16
  float*  bs = (float*)(As + 128 * 128);   // 128 f32

  int tid = threadIdx.x, lane = tid & 31, wave = tid >> 5;
  int r0 = blockIdx.x * 128;

  for (int x = tid; x < 128 * 128; x += 256) Ws[x] = to_bf16(W[x]);
  for (int x = tid; x < 128 * 128; x += 256) As[x] = to_bf16(A[(size_t)r0 * 128 + x]);
  if (tid < 128) bs[tid] = bias[tid];
  __syncthreads();

  f32x8 acc[8] = {};
  const __bf16* Arow = As + wave * 16 * 128;
#pragma unroll
  for (int ks = 0; ks < 128; ks += 32) {
    bf16x16 a = load_A_16x32(Arow, 128, ks, lane);
#pragma unroll
    for (int t = 0; t < 8; ++t) {
      bf16x16 bm = load_B_32x16(Ws, 128, ks, t * 16, lane);
      acc[t] = wmma_bf16(a, bm, acc[t]);
    }
  }

#pragma unroll
  for (int t = 0; t < 8; ++t) {
#pragma unroll
    for (int v = 0; v < 8; ++v) {
      int m = v + 8 * (lane >> 4);
      int n = (lane & 15) + 16 * t;
      int row = r0 + wave * 16 + m;
      float x = acc[t][v] + bs[n];
      if (act) x = softplusf(x);
      if (resid) x += resid[(size_t)row * 128 + n];
      Out[(size_t)row * 128 + n] = x;
    }
  }
}

// ---------------- fused SchNet interaction message kernel ----------------
// One workgroup per (b, target atom ti). Computes
//   agg[b,ti,f] = sum_j nm * ( sp(rbf@w1+b1) @ w2 + b2 )[j,f] * nf[b,j,f]
__global__ void __launch_bounds__(256)
msg_kernel(const float* __restrict__ pos,        // (B,N,3)
           const __bf16* __restrict__ w1bf,      // (64,NF)  bf16, this interaction
           const float* __restrict__ b1,         // (NF)
           const __bf16* __restrict__ w2bf,      // (NF,NF)  bf16
           const float* __restrict__ b2,         // (NF)
           const float* __restrict__ nf_g,       // (B,N,NF)
           const unsigned char* __restrict__ mask, // (B,N)
           float* __restrict__ agg)              // (B,N,NF)
{
  extern __shared__ char smem[];
  __bf16* w1_s  = (__bf16*)smem;                       // 64*128   = 16 KB
  __bf16* w2_s  = w1_s + NGP * NF;                     // 128*128  = 32 KB
  __bf16* rbf_s = w2_s + NF * NF;                      // 8*16*64  = 16 KB
  __bf16* H_s   = rbf_s + 8 * 16 * NGP;                // 8*16*128 = 32 KB
  float*  b1_s  = (float*)(H_s + 8 * 16 * NF);         // 128
  float*  b2_s  = b1_s + NF;                           // 128
  float*  agg_s = b2_s + NF;                           // 128
  float*  nm_s  = agg_s + NF;                          // 8*16

  int tid = threadIdx.x, lane = tid & 31, wave = tid >> 5;
  int b  = blockIdx.x >> 9;          // /512
  int ti = blockIdx.x & (NA - 1);

#if HAS_TDM
  // Tensor Data Mover: wave 0 DMAs the bf16 weight blobs straight into LDS.
  if (wave == 0) {
    tdm_load_bytes(w1bf, w1_s, NGP * NF * 2);
    tdm_load_bytes(w2bf, w2_s, NF * NF * 2);
    wait_tensorcnt0();
  }
#else
  for (int x = tid; x < (NGP * NF) / 8; x += 256)
    ((uint4*)w1_s)[x] = ((const uint4*)w1bf)[x];
  for (int x = tid; x < (NF * NF) / 8; x += 256)
    ((uint4*)w2_s)[x] = ((const uint4*)w2bf)[x];
#endif
  if (tid < NF) { b1_s[tid] = b1[tid]; b2_s[tid] = b2[tid]; agg_s[tid] = 0.0f; }
  __syncthreads();

  float px = pos[(b * NA + ti) * 3 + 0];
  float py = pos[(b * NA + ti) * 3 + 1];
  float pz = pos[(b * NA + ti) * 3 + 2];
  int   mi = mask[b * NA + ti];

  for (int p = 0; p < 4; ++p) {                 // 4 passes x 128 neighbours
    int jbase = p * 128 + wave * 16;            // this wave's 16 rows

    // ---- RBF for this wave's rows (each lane: 1 row x 32 gaussians) ----
    int r = lane & 15;
    int j = jbase + r;
    float dx = pos[(b * NA + j) * 3 + 0] - px;
    float dy = pos[(b * NA + j) * 3 + 1] - py;
    float dz = pos[(b * NA + j) * 3 + 2] - pz;
    float sq = dx * dx + dy * dy + dz * dz;
    float ds = sqrtf(sq > 0.0f ? sq : 1.0f);
    float d  = sq > 0.0f ? ds : 0.0f;
    int valid = mi && mask[b * NA + j] && (d <= CUTOFF_F) && (j != ti);
    if (lane < 16) nm_s[wave * 16 + r] = valid ? 1.0f : 0.0f;

    int gbase = (lane >> 4) * 32;
    uint32_t* myrbf32 = (uint32_t*)(rbf_s + (wave * 16 + r) * NGP + gbase);
#pragma unroll
    for (int gi = 0; gi < 32; gi += 2) {
      int g0i = gbase + gi;
      float v0 = 0.0f, v1 = 0.0f;
      if (valid) {
        if (g0i < NG) {
          float t0 = (d - (float)g0i * GSTEP) * (1.0f / GWIDTH);
          v0 = __expf(-0.5f * t0 * t0);
        }
        if (g0i + 1 < NG) {
          float t1 = (d - (float)(g0i + 1) * GSTEP) * (1.0f / GWIDTH);
          v1 = __expf(-0.5f * t1 * t1);
        }
      }
      myrbf32[gi >> 1] = (uint32_t)bf16_bits(v0) | ((uint32_t)bf16_bits(v1) << 16);
    }
    // produce/consume within the same wave -> compiler-inserted ds waits suffice

    // ---- GEMM1: (16 x 64) @ (64 x 128), bias, softplus -> H_s (bf16) ----
    {
      f32x8 acc[8] = {};
      const __bf16* Arow = rbf_s + wave * 16 * NGP;
#pragma unroll
      for (int ks = 0; ks < NGP; ks += 32) {
        bf16x16 a = load_A_16x32(Arow, NGP, ks, lane);
#pragma unroll
        for (int t = 0; t < 8; ++t) {
          bf16x16 bm = load_B_32x16(w1_s, NF, ks, t * 16, lane);
          acc[t] = wmma_bf16(a, bm, acc[t]);
        }
      }
#pragma unroll
      for (int t = 0; t < 8; ++t) {
#pragma unroll
        for (int v = 0; v < 8; ++v) {
          int m = v + 8 * (lane >> 4);
          int n = (lane & 15) + 16 * t;
          float x = softplusf(acc[t][v] + b1_s[n]);
          H_s[(wave * 16 + m) * NF + n] = to_bf16(x);
        }
      }
    }

    // ---- GEMM2: (16 x 128) @ (128 x 128), bias, * nm * nf, reduce over j ----
    {
      f32x8 acc[8] = {};
      const __bf16* Arow = H_s + wave * 16 * NF;
#pragma unroll
      for (int ks = 0; ks < NF; ks += 32) {
        bf16x16 a = load_A_16x32(Arow, NF, ks, lane);
#pragma unroll
        for (int t = 0; t < 8; ++t) {
          bf16x16 bm = load_B_32x16(w2_s, NF, ks, t * 16, lane);
          acc[t] = wmma_bf16(a, bm, acc[t]);
        }
      }
#pragma unroll
      for (int t = 0; t < 8; ++t) {
        int n = (lane & 15) + 16 * t;
        float sum = 0.0f;
#pragma unroll
        for (int v = 0; v < 8; ++v) {
          int m  = v + 8 * (lane >> 4);
          int jj = jbase + m;
          float x = (acc[t][v] + b2_s[n]) * nm_s[wave * 16 + m];
          sum += x * nf_g[((size_t)(b * NA + jj)) * NF + n];
        }
        atomicAdd(&agg_s[n], sum);   // ds_add_f32
      }
    }
  }

  __syncthreads();
  if (tid < NF) agg[((size_t)(b * NA + ti)) * NF + tid] = agg_s[tid];
}

// ---------------- masked per-atom energy + batch reduction ----------------
__global__ void __launch_bounds__(256)
energy_kernel(const float* __restrict__ h1, const float* __restrict__ aw2,
              const float* __restrict__ ab2, const unsigned char* __restrict__ mask,
              float* __restrict__ out) {
  __shared__ float w2s[HID];
  __shared__ float red[256];
  int b = blockIdx.x, tid = threadIdx.x;
  if (tid < HID) w2s[tid] = aw2[tid];
  __syncthreads();
  float accv = 0.0f;
  for (int i = tid; i < NA; i += 256) {
    const float* row = h1 + ((size_t)(b * NA + i)) * HID;
    float dot = 0.0f;
#pragma unroll 8
    for (int f = 0; f < HID; ++f) dot += row[f] * w2s[f];
    if (mask[b * NA + i]) accv += dot + ab2[0];
  }
  red[tid] = accv;
  __syncthreads();
  for (int s = 128; s > 0; s >>= 1) {
    if (tid < s) red[tid] += red[tid + s];
    __syncthreads();
  }
  if (tid == 0) out[b] = red[0];
}

// ---------------- host-side orchestration ----------------
extern "C" void kernel_launch(void* const* d_in, const int* in_sizes, int n_in,
                              void* d_out, int out_size, void* d_ws, size_t ws_size,
                              hipStream_t stream) {
  const float* pos     = (const float*)d_in[0];
  const float* emb     = (const float*)d_in[1];
  const float* rbf_w1  = (const float*)d_in[2];
  const float* rbf_b1  = (const float*)d_in[3];
  const float* rbf_w2  = (const float*)d_in[4];
  const float* rbf_b2  = (const float*)d_in[5];
  const float* f_w     = (const float*)d_in[6];
  const float* f_b     = (const float*)d_in[7];
  const float* out_w1  = (const float*)d_in[8];
  const float* out_b1  = (const float*)d_in[9];
  const float* out_w2  = (const float*)d_in[10];
  const float* out_b2  = (const float*)d_in[11];
  const float* atom_w1 = (const float*)d_in[12];
  const float* atom_b1 = (const float*)d_in[13];
  const float* atom_w2 = (const float*)d_in[14];
  const float* atom_b2 = (const float*)d_in[15];
  const int*   node_idx = (const int*)d_in[16];
  const unsigned char* mask = (const unsigned char*)d_in[17];
  float* out = (float*)d_out;

  float* feats = (float*)d_ws;                 // (B*N, 128) f32
  float* nf    = feats + (size_t)NB * NA * HID;
  float* agg   = nf    + (size_t)NB * NA * NF;
  float* tmp   = agg   + (size_t)NB * NA * NF;
  __bf16* w1bf = (__bf16*)(tmp + (size_t)NB * NA * HID);  // NI * 64*128 bf16
  __bf16* w2bf = w1bf + (size_t)NI * NGP * NF;            // NI * 128*128 bf16

  const int M = NB * NA;                       // 2048 rows
  int total = M * HID;
  embed_kernel<<<(total + 255) / 256, 256, 0, stream>>>(emb, node_idx, feats, total);

  int packTotal = NI * NGP * NF + NI * NF * NF;
  pack_weights_kernel<<<(packTotal + 255) / 256, 256, 0, stream>>>(
      rbf_w1, rbf_w2, w1bf, w2bf);

  size_t gemmSmem = (size_t)(2 * 128 * 128) * sizeof(__bf16) + 128 * sizeof(float);
  size_t msgSmem  = (size_t)(NGP * NF + NF * NF + 8 * 16 * NGP + 8 * 16 * NF) * sizeof(__bf16)
                  + (size_t)(3 * NF + 8 * 16) * sizeof(float);

  for (int it = 0; it < NI; ++it) {
    // nf = feats @ f_w[it] + f_b[it]
    gemm128_kernel<<<M / 128, 256, gemmSmem, stream>>>(
        feats, f_w + (size_t)it * HID * NF, f_b + (size_t)it * NF,
        nullptr, nf, M, 0);
    // agg = fused pairwise message passing
    msg_kernel<<<NB * NA, 256, msgSmem, stream>>>(
        pos, w1bf + (size_t)it * NGP * NF, rbf_b1 + (size_t)it * NF,
        w2bf + (size_t)it * NF * NF, rbf_b2 + (size_t)it * NF,
        nf, mask, agg);
    // tmp = softplus(agg @ out_w1 + out_b1)
    gemm128_kernel<<<M / 128, 256, gemmSmem, stream>>>(
        agg, out_w1 + (size_t)it * NF * HID, out_b1 + (size_t)it * HID,
        nullptr, tmp, M, 1);
    // feats = feats + (tmp @ out_w2 + out_b2)
    gemm128_kernel<<<M / 128, 256, gemmSmem, stream>>>(
        tmp, out_w2 + (size_t)it * HID * HID, out_b2 + (size_t)it * HID,
        feats, feats, M, 0);
  }

  // atom MLP hidden: tmp = softplus(feats @ atom_w1 + atom_b1)
  gemm128_kernel<<<M / 128, 256, gemmSmem, stream>>>(
      feats, atom_w1, atom_b1, nullptr, tmp, M, 1);
  // per-atom energy, mask, batch-sum
  energy_kernel<<<NB, 256, 0, stream>>>(tmp, atom_w2, atom_b2, mask, out);
}
